// MultiHeadAttention_41480794145038
// MI455X (gfx1250) — compile-verified, matrix operand order checked by reading, not run on hardware
//
#include <hip/hip_runtime.h>
#include <hip/hip_bf16.h>

// ---------------------------------------------------------------------------
// MHA for MI455X (gfx1250): bf16 WMMA GEMMs + flash attention, f32 accum.
// B=2, S=2048, E=1024, H=16, D=64, softmax scale = 1/8.
// Uses global_load_async_to_lds_b128 (ASYNCcnt) for LDS staging with
// double-buffered GEMM tiles.
// ---------------------------------------------------------------------------

typedef __attribute__((ext_vector_type(16))) __bf16 v16bf;
typedef __attribute__((ext_vector_type(8)))  float  v8f;

union Frag16 {
    v16bf v;
    uint4 q[2];
    __bf16 e[16];
};
union Chunk32 {
    uint4 q[4];
    __bf16 e[32];
};

static __device__ __forceinline__ __bf16 f2bf(float f) {
    union { float f; unsigned u; } a; a.f = f;
    unsigned r = a.u + 0x7FFFu + ((a.u >> 16) & 1u);   // round-to-nearest-even
    union { unsigned short s; __bf16 b; } o; o.s = (unsigned short)(r >> 16);
    return o.b;
}

// Generic (flat) pointer to LDS byte offset: LDS aperture maps addr[31:0].
static __device__ __forceinline__ unsigned lds_off(const void* p) {
    return (unsigned)(size_t)p;
}

// Load a 16x32 (A: MxK) or 32x16-as-columns (B: KxN) bf16 fragment.
// Layout per ISA 7.12.2: lane<16 -> row=lane, K 0..7 & 16..23;
//                        lane>=16 -> row=lane-16, K 8..15 & 24..31.
static __device__ __forceinline__ v16bf load_frag(const __bf16* base, int stride) {
    const int lane = threadIdx.x & 31;
    const int r    = lane & 15;
    const int kh   = (lane >> 4) * 8;
    const __bf16* p = base + (size_t)r * stride + kh;
    Frag16 f;
    f.q[0] = *(const uint4*)(p);        // K kh .. kh+7
    f.q[1] = *(const uint4*)(p + 16);   // K kh+16 .. kh+23
    return f.v;
}

static __device__ __forceinline__ v8f wmma_bf16(v16bf a, v16bf b, v8f c) {
    return __builtin_amdgcn_wmma_f32_16x16x32_bf16(
        /*neg_a=*/false, a, /*neg_b=*/false, b,
        /*c_mod=*/(short)0, c, /*reuse_a=*/false, /*reuse_b=*/false);
}

// ---------------------------------------------------------------------------
// fp32 -> bf16 conversion
// ---------------------------------------------------------------------------
__global__ void cvt_f32_bf16(const float* __restrict__ in,
                             __bf16* __restrict__ out, int n) {
    int i = blockIdx.x * blockDim.x + threadIdx.x;
    if (i < n) out[i] = f2bf(in[i]);
}

// ---------------------------------------------------------------------------
// C[M,N] = A[M,K] @ W[N,K]^T + bias   (torch Linear semantics)
// A, W bf16 row-major. out_mode 0: f32 [M,N]. out_mode 1: bf16 [B,H,S,D]
// with row = b*S+s (token), col = h*D+d.
// Block: 256 threads = 8 waves, 128x128 tile; wave = 32x64 sub-tile.
// K tiles (128x32 bf16) staged via async-to-LDS, double buffered.
// ---------------------------------------------------------------------------
#define GK 1024
__global__ __launch_bounds__(256) void gemm_bf16(
    const __bf16* __restrict__ A, const __bf16* __restrict__ W,
    const float* __restrict__ bias, void* __restrict__ Out,
    int M, int N, int out_mode) {

    __shared__ __bf16 As[2][128][40];   // 128 rows x 32 k, padded, 2 buffers
    __shared__ __bf16 Ws[2][128][40];

    const int m0 = blockIdx.y * 128;
    const int n0 = blockIdx.x * 128;
    const int t  = threadIdx.x;
    const int lane = t & 31;
    const int wave = t >> 5;
    const int m_base = (wave & 3) * 32;   // 4 waves along M
    const int n_base = (wave >> 2) * 64;  // 2 waves along N

    const int lr = t >> 1;            // 0..127: staged row
    const int lc = (t & 1) * 16;      // k-chunk of 16 halves (32 B)

    // issue 4 async b128 loads: 32 B of As-tile + 32 B of Ws-tile per thread.
    // offset: applies to BOTH the LDS and global address (ISA 8.2).
    auto stage = [&](int buf, int k0) {
        const unsigned  la = lds_off(&As[buf][lr][lc]);
        const unsigned  lw = lds_off(&Ws[buf][lr][lc]);
        const unsigned long long ga =
            (unsigned long long)(size_t)(A + (size_t)(m0 + lr) * GK + k0 + lc);
        const unsigned long long gw =
            (unsigned long long)(size_t)(W + (size_t)(n0 + lr) * GK + k0 + lc);
        asm volatile(
            "global_load_async_to_lds_b128 %0, %2, off\n\t"
            "global_load_async_to_lds_b128 %0, %2, off offset:16\n\t"
            "global_load_async_to_lds_b128 %1, %3, off\n\t"
            "global_load_async_to_lds_b128 %1, %3, off offset:16"
            :: "v"(la), "v"(lw), "v"(ga), "v"(gw) : "memory");
    };

    v8f acc[2][4] = {};

    stage(0, 0);
    for (int k0 = 0; k0 < GK; k0 += 32) {
        const int buf = (k0 >> 5) & 1;
        if (k0 + 32 < GK) {
            stage(buf ^ 1, k0 + 32);
            // 4 just-issued may be pending; previous tile's 4 are complete.
            asm volatile("s_wait_asynccnt 0x4" ::: "memory");
        } else {
            asm volatile("s_wait_asynccnt 0x0" ::: "memory");
        }
        __syncthreads();

        v16bf af[2], bf[4];
        #pragma unroll
        for (int i = 0; i < 2; ++i) af[i] = load_frag(&As[buf][m_base + i * 16][0], 40);
        #pragma unroll
        for (int j = 0; j < 4; ++j) bf[j] = load_frag(&Ws[buf][n_base + j * 16][0], 40);
        #pragma unroll
        for (int i = 0; i < 2; ++i)
            #pragma unroll
            for (int j = 0; j < 4; ++j)
                acc[i][j] = wmma_bf16(af[i], bf[j], acc[i][j]);
        __syncthreads();
    }

    // Epilogue. C/D layout: VGPR v, lanes 0-15 -> row v, lanes 16-31 -> row v+8.
    const int rhalf = (lane >> 4) * 8;
    const int cl    = lane & 15;
    #pragma unroll
    for (int i = 0; i < 2; ++i) {
        #pragma unroll
        for (int j = 0; j < 4; ++j) {
            const int col = n0 + n_base + j * 16 + cl;
            const float bv = bias[col];
            #pragma unroll
            for (int v = 0; v < 8; ++v) {
                const int row = m0 + m_base + i * 16 + v + rhalf;
                const float val = acc[i][j][v] + bv;
                if (out_mode == 0) {
                    ((float*)Out)[(size_t)row * N + col] = val;
                } else {
                    // [B,H,S,D]: b=row>>11, s=row&2047, h=col>>6, d=col&63
                    const size_t idx = ((((size_t)(row >> 11) * 16) + (col >> 6)) * 2048
                                        + (row & 2047)) * 64 + (col & 63);
                    ((__bf16*)Out)[idx] = f2bf(val);
                }
            }
        }
    }
}

// ---------------------------------------------------------------------------
// Flash attention. Q,K,V bf16 [B,H,S,D] (D=64). Out bf16 token-major
// [B*S, E] with col = h*64+d. Grid: (S/64, B*H). Block: 128 thr = 4 waves,
// each wave owns 16 query rows. Key tiles of 64, online softmax in f32.
// K tile staged via async-to-LDS; V tile transposed through registers.
// ---------------------------------------------------------------------------
__global__ __launch_bounds__(128) void flash_attn(
    const __bf16* __restrict__ Q, const __bf16* __restrict__ Kb,
    const __bf16* __restrict__ Vb, __bf16* __restrict__ Ob) {

    __shared__ __bf16 Ks[64][72];        // key tile, row-major [t][d]
    __shared__ __bf16 Vt[64][72];        // value tile, transposed [d][t]
    __shared__ __bf16 Ps[4][16][72];     // per-wave P scratch [row][t]

    const int qt   = blockIdx.x;
    const int bh   = blockIdx.y;
    const int t    = threadIdx.x;
    const int lane = t & 31;
    const int wave = t >> 5;

    const size_t head_off = (size_t)bh * 2048 * 64;

    // Q fragments (16 rows x 64 d) straight from global, per ISA A layout.
    const __bf16* qbase = Q + head_off + (size_t)(qt * 64 + wave * 16) * 64;
    v16bf qf[2];
    #pragma unroll
    for (int k = 0; k < 2; ++k) qf[k] = load_frag(qbase + k * 32, 64);

    float mrow[8], lrow[8];
    #pragma unroll
    for (int v = 0; v < 8; ++v) { mrow[v] = -__builtin_inff(); lrow[v] = 0.f; }
    v8f oacc[4] = {};

    const int lr = t >> 1;            // 0..63
    const int lc = (t & 1) * 32;      // 0 or 32 halves (64 B)

    for (int kt = 0; kt < 32; ++kt) {
        // K tile: async global -> LDS (64 B per thread, contiguous in-row)
        {
            const unsigned lk = lds_off(&Ks[lr][lc]);
            const unsigned long long gk = (unsigned long long)(size_t)
                (Kb + head_off + (size_t)(kt * 64 + lr) * 64 + lc);
            asm volatile(
                "global_load_async_to_lds_b128 %0, %1, off\n\t"
                "global_load_async_to_lds_b128 %0, %1, off offset:16\n\t"
                "global_load_async_to_lds_b128 %0, %1, off offset:32\n\t"
                "global_load_async_to_lds_b128 %0, %1, off offset:48"
                :: "v"(lk), "v"(gk) : "memory");
        }
        // V tile: transpose through registers into Vt[d][t]
        {
            Chunk32 cv;
            const __bf16* gv = Vb + head_off + (size_t)(kt * 64 + lr) * 64 + lc;
            #pragma unroll
            for (int q = 0; q < 4; ++q) cv.q[q] = *(const uint4*)(gv + q * 8);
            #pragma unroll
            for (int q = 0; q < 32; ++q) Vt[lc + q][lr] = cv.e[q];
        }
        asm volatile("s_wait_asynccnt 0x0" ::: "memory");
        __syncthreads();

        // S = Q @ K^T  (4 key-column tiles x 2 K-steps over d)
        v8f sacc[4] = {};
        #pragma unroll
        for (int k = 0; k < 2; ++k) {
            #pragma unroll
            for (int n = 0; n < 4; ++n) {
                v16bf bfb = load_frag(&Ks[n * 16][k * 32], 72);
                sacc[n] = wmma_bf16(qf[k], bfb, sacc[n]);
            }
        }

        // online softmax (scale 1/8)
        #pragma unroll
        for (int v = 0; v < 8; ++v) {
            float s0 = sacc[0][v] * 0.125f, s1 = sacc[1][v] * 0.125f;
            float s2 = sacc[2][v] * 0.125f, s3 = sacc[3][v] * 0.125f;
            sacc[0][v] = s0; sacc[1][v] = s1; sacc[2][v] = s2; sacc[3][v] = s3;
            float mx = fmaxf(fmaxf(s0, s1), fmaxf(s2, s3));
            #pragma unroll
            for (int off = 8; off >= 1; off >>= 1)
                mx = fmaxf(mx, __shfl_xor(mx, off, 32));
            const float mnew  = fmaxf(mrow[v], mx);
            const float alpha = __expf(mrow[v] - mnew);
            mrow[v] = mnew;
            float rs = 0.f;
            #pragma unroll
            for (int n = 0; n < 4; ++n) {
                const float p = __expf(sacc[n][v] - mnew);
                sacc[n][v] = p;
                rs += p;
            }
            #pragma unroll
            for (int off = 8; off >= 1; off >>= 1)
                rs += __shfl_xor(rs, off, 32);
            lrow[v] = lrow[v] * alpha + rs;
            #pragma unroll
            for (int j = 0; j < 4; ++j) oacc[j][v] *= alpha;
        }

        // P -> wave-local LDS (bf16), then O += P @ V
        {
            const int rh = (lane >> 4) * 8, cl = lane & 15;
            #pragma unroll
            for (int n = 0; n < 4; ++n)
                #pragma unroll
                for (int v = 0; v < 8; ++v)
                    Ps[wave][v + rh][n * 16 + cl] = f2bf(sacc[n][v]);
        }
        asm volatile("s_wait_dscnt 0x0" ::: "memory");   // wave-local LDS RAW

        #pragma unroll
        for (int k = 0; k < 2; ++k) {
            v16bf pa = load_frag(&Ps[wave][0][k * 32], 72);
            #pragma unroll
            for (int j = 0; j < 4; ++j) {
                v16bf vb = load_frag(&Vt[j * 16][k * 32], 72);
                oacc[j] = wmma_bf16(pa, vb, oacc[j]);
            }
        }
        __syncthreads();
    }

    // normalize and write token-major bf16 [token][h*64+d]
    const int b = bh >> 4, h = bh & 15;
    const int rh = (lane >> 4) * 8, cl = lane & 15;
    #pragma unroll
    for (int v = 0; v < 8; ++v) {
        const float inv = 1.0f / lrow[v];
        const int srow  = qt * 64 + wave * 16 + v + rh;
        const size_t row = (size_t)b * 2048 + srow;
        #pragma unroll
        for (int j = 0; j < 4; ++j)
            Ob[row * 1024 + h * 64 + j * 16 + cl] = f2bf(oacc[j][v] * inv);
    }
}

// ---------------------------------------------------------------------------
extern "C" void kernel_launch(void* const* d_in, const int* in_sizes, int n_in,
                              void* d_out, int out_size, void* d_ws, size_t ws_size,
                              hipStream_t stream) {
    const float* x  = (const float*)d_in[0];
    const float* Wq = (const float*)d_in[1];
    const float* bq = (const float*)d_in[2];
    const float* Wk = (const float*)d_in[3];
    const float* bk = (const float*)d_in[4];
    const float* Wv = (const float*)d_in[5];
    const float* bv = (const float*)d_in[6];
    const float* Wo = (const float*)d_in[7];
    const float* bo = (const float*)d_in[8];

    const int M = 4096;              // B*S tokens
    const int E = 1024;
    char* ws = (char*)d_ws;
    const size_t MB = 1024 * 1024;
    __bf16* Xb  = (__bf16*)(ws + 0 * MB);    // 8 MB
    __bf16* Wqb = (__bf16*)(ws + 8 * MB);    // 2 MB
    __bf16* Wkb = (__bf16*)(ws + 10 * MB);
    __bf16* Wvb = (__bf16*)(ws + 12 * MB);
    __bf16* Wob = (__bf16*)(ws + 14 * MB);
    __bf16* Qb  = (__bf16*)(ws + 16 * MB);   // 8 MB each, [B,H,S,D]
    __bf16* Kb  = (__bf16*)(ws + 24 * MB);
    __bf16* Vb  = (__bf16*)(ws + 32 * MB);
    __bf16* Ab  = (__bf16*)(ws + 40 * MB);   // attention out, token-major

    // fp32 -> bf16
    cvt_f32_bf16<<<(M * E + 255) / 256, 256, 0, stream>>>(x, Xb, M * E);
    cvt_f32_bf16<<<(E * E + 255) / 256, 256, 0, stream>>>(Wq, Wqb, E * E);
    cvt_f32_bf16<<<(E * E + 255) / 256, 256, 0, stream>>>(Wk, Wkb, E * E);
    cvt_f32_bf16<<<(E * E + 255) / 256, 256, 0, stream>>>(Wv, Wvb, E * E);
    cvt_f32_bf16<<<(E * E + 255) / 256, 256, 0, stream>>>(Wo, Wob, E * E);

    // Q/K/V projections -> bf16 head layout
    dim3 ggrid(E / 128, M / 128);
    gemm_bf16<<<ggrid, 256, 0, stream>>>(Xb, Wqb, bq, Qb, M, E, 1);
    gemm_bf16<<<ggrid, 256, 0, stream>>>(Xb, Wkb, bk, Kb, M, E, 1);
    gemm_bf16<<<ggrid, 256, 0, stream>>>(Xb, Wvb, bv, Vb, M, E, 1);

    // flash attention
    flash_attn<<<dim3(2048 / 64, 32), 128, 0, stream>>>(Qb, Kb, Vb, Ab);

    // output projection -> f32 d_out
    gemm_bf16<<<ggrid, 256, 0, stream>>>(Ab, Wob, bo, d_out, M, E, 0);
}
